// GroupedQueryAttention_63101659513095
// MI455X (gfx1250) — matrix operand order repaired and checked by reading, not compiled, not run on hardware
//
#include <hip/hip_runtime.h>
#include <hip/hip_bf16.h>

typedef unsigned short u16;
typedef unsigned int   u32;

typedef __attribute__((ext_vector_type(16))) __bf16 v16bf;
typedef __attribute__((ext_vector_type(8)))  float  v8f;

#define B_   2
#define S_   2048
#define D_   1024
#define H_   16
#define HKV_ 4
#define HD_  64
#define NREP_ 4

__device__ __forceinline__ u16 f32_bf16(float f) {
  union { float f; u32 u; } c{f};
  u32 lsb = (c.u >> 16) & 1u;
  return (u16)((c.u + 0x7FFFu + lsb) >> 16);
}

// A-matrix (16x32 bf16) K index for VGPR pair p, lane-half hl (ISA 7.12.2)
__device__ __forceinline__ int fkA(int p, int hl) {
  return ((p & 3) * 2) + ((p & 4) ? 16 : 0) + hl * 8;
}
// B-matrix (32x16 bf16) K index for VGPR pair p, lane-half hl (same K order as A)
__device__ __forceinline__ int fkB(int p, int hl) {
  return ((p & 3) * 2) + ((p & 4) ? 16 : 0) + hl * 8;
}

__device__ __forceinline__ v8f v8zero() {
  v8f v = {0.f, 0.f, 0.f, 0.f, 0.f, 0.f, 0.f, 0.f};
  return v;
}

__device__ __forceinline__ u32 lds_off(const void* p) {
  // flat LDS aperture: addr[31:0] is the LDS byte offset
  return (u32)(size_t)p;
}

// ---------------------------------------------------------------------------
// fp32 -> bf16 elementwise (for x / activations)
// ---------------------------------------------------------------------------
__global__ void cvt_bf16(const float* __restrict__ in, u16* __restrict__ out, int n) {
  int i = blockIdx.x * blockDim.x + threadIdx.x;
  int stride = gridDim.x * blockDim.x;
  for (; i < n; i += stride) out[i] = f32_bf16(in[i]);
}

// ---------------------------------------------------------------------------
// fp32 [R x C] -> bf16 transposed [C x R]  (weights; tiled through LDS)
// grid (C/32, R/32), block (32, 8)
// ---------------------------------------------------------------------------
__global__ __launch_bounds__(256) void cvt_bf16_T(const float* __restrict__ in,
                                                  u16* __restrict__ out,
                                                  int R, int Ccol) {
  __shared__ u16 t[32][33];
  const int bx = blockIdx.x * 32;  // input col base
  const int by = blockIdx.y * 32;  // input row base
  const int tx = threadIdx.x;
  const int ty = threadIdx.y;
#pragma unroll
  for (int i = 0; i < 32; i += 8)
    t[ty + i][tx] = f32_bf16(in[(size_t)(by + ty + i) * Ccol + bx + tx]);
  __syncthreads();
#pragma unroll
  for (int i = 0; i < 32; i += 8)
    out[(size_t)(bx + ty + i) * R + by + tx] = t[tx][ty + i];
}

// ---------------------------------------------------------------------------
// bf16 GEMM: C[MxN] (f32) = A[MxK] * Bt[NxK]^T, K fixed = 1024.
// Block: 256 threads = 8 waves (2x4), block tile 64x256, wave tile 32x64.
// Double-buffered LDS; tiles staged with GLOBAL_LOAD_ASYNC_TO_LDS_B128 and
// pipelined on ASYNCcnt (issue t+1, wait <=5 so tile t is resident).
// ---------------------------------------------------------------------------
#define BM 64
#define BN 256
#define BK 32
#define GK 1024
#define NTILE (GK / BK)
#define LDA 40   // u16 row stride (80 B, 16-B aligned)
#define LDBT 40

__global__ __launch_bounds__(256) void gemm_bf16_f32(const u16* __restrict__ A,
                                                     const u16* __restrict__ Bt,
                                                     float* __restrict__ C,
                                                     int N) {
  __shared__ u16 sA[2][BM * LDA];    // 2 x 5120 B
  __shared__ u16 sB[2][BN * LDBT];   // 2 x 20480 B
  const int tid  = threadIdx.x;
  const int wv   = tid >> 5;
  const int lane = tid & 31;
  const int n0   = lane & 15;
  const int hl   = lane >> 4;
  const int wm   = (wv >> 2) * 32;
  const int wn   = (wv & 3) * 64;

  const u16* Ag = A  + (size_t)(blockIdx.x * BM) * GK;
  const u16* Bg = Bt + (size_t)(blockIdx.y * BN) * GK;

  // staging coords: one b128 (8 u16) per thread for A, four for B
  const int ar = tid >> 2;
  const int ac = (tid & 3) * 8;

  auto issue = [&](int t, int buf) {
    const int k0 = t * BK;
    {
      u32 la = lds_off(&sA[buf][ar * LDA + ac]);
      const u16* ga = Ag + ar * GK + k0 + ac;
      asm volatile("global_load_async_to_lds_b128 %0, %1, off"
                   :: "v"(la), "v"(ga) : "memory");
    }
#pragma unroll
    for (int i = 0; i < 4; ++i) {
      int idx = tid + i * 256;
      int r = idx >> 2, c = (idx & 3) * 8;
      u32 lb = lds_off(&sB[buf][r * LDBT + c]);
      const u16* gb = Bg + r * GK + k0 + c;
      asm volatile("global_load_async_to_lds_b128 %0, %1, off"
                   :: "v"(lb), "v"(gb) : "memory");
    }
  };

  v8f acc[2][4];
#pragma unroll
  for (int i = 0; i < 2; ++i)
#pragma unroll
    for (int j = 0; j < 4; ++j) acc[i][j] = v8zero();

  issue(0, 0);

#pragma unroll 2
  for (int t = 0; t < NTILE; ++t) {
    const int cur = t & 1;
    if (t + 1 < NTILE) {
      issue(t + 1, cur ^ 1);
      asm volatile("s_wait_asynccnt 0x5" ::: "memory");  // tile t resident
    } else {
      asm volatile("s_wait_asynccnt 0x0" ::: "memory");
    }
    __syncthreads();

    v16bf af[2];
#pragma unroll
    for (int i = 0; i < 2; ++i) {
      union { v16bf v; u32 u[8]; } f;
      const u16* base = &sA[cur][(wm + i * 16 + n0) * LDA];
#pragma unroll
      for (int p = 0; p < 8; ++p) f.u[p] = *(const u32*)(base + fkA(p, hl));
      af[i] = f.v;
    }
    v16bf bfr[4];
#pragma unroll
    for (int j = 0; j < 4; ++j) {
      union { v16bf v; u32 u[8]; } f;
      const u16* base = &sB[cur][(wn + j * 16 + n0) * LDBT];
#pragma unroll
      for (int p = 0; p < 8; ++p) f.u[p] = *(const u32*)(base + fkB(p, hl));
      bfr[j] = f.v;
    }
#pragma unroll
    for (int i = 0; i < 2; ++i)
#pragma unroll
      for (int j = 0; j < 4; ++j)
        acc[i][j] = __builtin_amdgcn_wmma_f32_16x16x32_bf16(
            false, af[i], false, bfr[j], (short)0, acc[i][j], false, false);
    __syncthreads();   // all waves done reading buf[cur] before it is re-filled
  }

  // epilogue: 64-bit base hoisted once, 32-bit offsets per store
  float* Cb = C + (size_t)(blockIdx.x * BM + wm) * N + blockIdx.y * BN + wn;
#pragma unroll
  for (int i = 0; i < 2; ++i)
#pragma unroll
    for (int j = 0; j < 4; ++j)
#pragma unroll
      for (int r = 0; r < 8; ++r)
        Cb[(i * 16 + r + hl * 8) * N + j * 16 + n0] = acc[i][j][r];
}

// ---------------------------------------------------------------------------
// RoPE + repack [B*S, nh*HD] f32 -> [B, nh, S, HD] bf16 (scaled)
// ---------------------------------------------------------------------------
__global__ void rope_pack(const float* __restrict__ in, u16* __restrict__ out,
                          int nh, float scale) {
  int idx = blockIdx.x * blockDim.x + threadIdx.x;  // [B, nh, S, HD] linear
  int d  = idx & (HD_ - 1);
  int t1 = idx >> 6;
  int s  = t1 & (S_ - 1);
  int t2 = t1 >> 11;
  int h  = t2 % nh;
  int b  = t2 / nh;
  const float* row = in + (size_t)(b * S_ + s) * ((size_t)nh * HD_) + h * HD_;
  float x1 = row[d];
  int i = d & 31;
  float inv = __expf(-(float)i * (9.210340371976184f / 32.0f)); // 10000^(-i/32)
  float ang = (float)s * inv;
  float sn, cs;
  __sincosf(ang, &sn, &cs);
  float x2 = (d < 32) ? -row[d + 32] : row[d - 32];
  out[idx] = f32_bf16((x1 * cs + x2 * sn) * scale);
}

// V: [B*S, HKV*HD] f32 -> transposed [B, HKV, HD, S] bf16 (contiguous PV B-frags)
__global__ void convert_v_T(const float* __restrict__ in, u16* __restrict__ out) {
  int idx = blockIdx.x * blockDim.x + threadIdx.x;  // [B, HKV, HD, S] linear
  int s  = idx & (S_ - 1);
  int t  = idx >> 11;
  int d  = t & (HD_ - 1);
  int t2 = t >> 6;
  int h  = t2 & (HKV_ - 1);
  int b  = t2 >> 2;
  float v = in[(size_t)(b * S_ + s) * (HKV_ * HD_) + h * HD_ + d];
  out[idx] = f32_bf16(v);
}

// ---------------------------------------------------------------------------
// K^T B-fragment loader: 32-key tile -> 4 frags (2 n-tiles x 2 k-steps)
// ---------------------------------------------------------------------------
__device__ __forceinline__ void load_kfrags(const u16* __restrict__ Kp, int kt,
                                            int n0, int hl, v16bf out[4]) {
#pragma unroll
  for (int nt = 0; nt < 2; ++nt) {
    union { v16bf v; u32 u[8]; } f0, f1;
    const u16* kr = Kp + (kt + nt * 16 + n0) * HD_;
#pragma unroll
    for (int p = 0; p < 8; ++p) {
      f0.u[p] = *(const u32*)(kr + fkB(p, hl));
      f1.u[p] = *(const u32*)(kr + 32 + fkB(p, hl));
    }
    out[nt * 2 + 0] = f0.v;
    out[nt * 2 + 1] = f1.v;
  }
}

// ---------------------------------------------------------------------------
// Flash attention. Block = 128 threads (4 waves), each wave owns 16 q-rows.
// Per iteration: 32 keys -> QK^T (4 wmma), online softmax, P.V (4 wmma).
// K frags prefetched one tile ahead; V frags loaded early (hide under softmax);
// P staging double-buffered -> one barrier per tile.
// ---------------------------------------------------------------------------
__global__ __launch_bounds__(128) void attn_fwd(const u16* __restrict__ Q,
                                                const u16* __restrict__ Kc,
                                                const u16* __restrict__ Vt,
                                                u16* __restrict__ O) {
  __shared__ u16 pst[4][2][16 * 40];   // per-wave, double-buffered P staging
  const int wv   = threadIdx.x >> 5;
  const int lane = threadIdx.x & 31;
  const int n0   = lane & 15;
  const int hl   = lane >> 4;

  const int qpb  = S_ / 64;
  const int bh   = blockIdx.x / qpb;
  const int qblk = blockIdx.x % qpb;
  const int b = bh / H_, h = bh % H_;
  const int hkv = h / NREP_;
  const int qbase = qblk * 64 + wv * 16;

  const u16* Qp = Q  + (size_t)((b * H_   + h)   * S_)  * HD_;
  const u16* Kp = Kc + (size_t)((b * HKV_ + hkv) * S_)  * HD_;
  const u16* Vp = Vt + (size_t)((b * HKV_ + hkv) * HD_) * S_;

  v16bf aq[2];
#pragma unroll
  for (int ks = 0; ks < 2; ++ks) {
    union { v16bf v; u32 u[8]; } f;
#pragma unroll
    for (int p = 0; p < 8; ++p)
      f.u[p] = *(const u32*)(Qp + (qbase + n0) * HD_ + ks * 32 + fkA(p, hl));
    aq[ks] = f.v;
  }

  v8f o[4];
#pragma unroll
  for (int t = 0; t < 4; ++t) o[t] = v8zero();
  float mrow[8], lrow[8];
#pragma unroll
  for (int r = 0; r < 8; ++r) { mrow[r] = -1e30f; lrow[r] = 0.f; }

  v16bf kf[2][4];
  load_kfrags(Kp, 0, n0, hl, kf[0]);

#pragma unroll 2
  for (int kt = 0; kt < S_; kt += 32) {
    const int cur = (kt >> 5) & 1;

    // ---- scores = Q * K^T ----
    v8f sc[2];
#pragma unroll
    for (int nt = 0; nt < 2; ++nt) {
      sc[nt] = __builtin_amdgcn_wmma_f32_16x16x32_bf16(
          false, aq[0], false, kf[cur][nt * 2 + 0], (short)0, v8zero(), false, false);
      sc[nt] = __builtin_amdgcn_wmma_f32_16x16x32_bf16(
          false, aq[1], false, kf[cur][nt * 2 + 1], (short)0, sc[nt], false, false);
    }

    // ---- prefetch next K tile ----
    if (kt + 32 < S_) load_kfrags(Kp, kt + 32, n0, hl, kf[cur ^ 1]);

    // ---- early V frag loads (latency hidden under softmax + staging) ----
    v16bf fv[4];
#pragma unroll
    for (int t = 0; t < 4; ++t) {
      union { v16bf v; u32 u[8]; } f;
      const u16* vr = Vp + (t * 16 + n0) * S_ + kt;
#pragma unroll
      for (int p = 0; p < 8; ++p) f.u[p] = *(const u32*)(vr + fkB(p, hl));
      fv[t] = f.v;
    }

    // ---- online softmax (rows live within a 16-lane half) ----
    float alpha[8];
#pragma unroll
    for (int r = 0; r < 8; ++r) {
      float mx = fmaxf(sc[0][r], sc[1][r]);
#pragma unroll
      for (int msk = 8; msk; msk >>= 1) mx = fmaxf(mx, __shfl_xor(mx, msk, 32));
      float mn = fmaxf(mrow[r], mx);
      float al = __expf(mrow[r] - mn);
      float p0 = __expf(sc[0][r] - mn);
      float p1 = __expf(sc[1][r] - mn);
      sc[0][r] = p0; sc[1][r] = p1;
      float ps = p0 + p1;
#pragma unroll
      for (int msk = 8; msk; msk >>= 1) ps += __shfl_xor(ps, msk, 32);
      lrow[r] = lrow[r] * al + ps;
      mrow[r] = mn;
      alpha[r] = al;
    }
#pragma unroll
    for (int t = 0; t < 4; ++t)
#pragma unroll
      for (int r = 0; r < 8; ++r) o[t][r] *= alpha[r];

    // ---- C-layout -> A-layout for P via double-buffered LDS ----
    u16* pw = &pst[wv][cur][0];
#pragma unroll
    for (int nt = 0; nt < 2; ++nt)
#pragma unroll
      for (int r = 0; r < 8; ++r)
        pw[(r + hl * 8) * 40 + nt * 16 + n0] = f32_bf16(sc[nt][r]);
    __syncthreads();
    union { v16bf v; u32 u[8]; } fp;
#pragma unroll
    for (int p = 0; p < 8; ++p)
      fp.u[p] = *(const u32*)(pw + n0 * 40 + fkA(p, hl));

    // ---- O += P * V ----
#pragma unroll
    for (int t = 0; t < 4; ++t)
      o[t] = __builtin_amdgcn_wmma_f32_16x16x32_bf16(
          false, fp.v, false, fv[t], (short)0, o[t], false, false);
  }

  // ---- finalize: O /= l, write [B*S, D] bf16 ----
  u16* Op = O + (size_t)(b * S_ + qbase) * D_ + h * HD_;
#pragma unroll
  for (int t = 0; t < 4; ++t)
#pragma unroll
    for (int r = 0; r < 8; ++r)
      Op[(r + hl * 8) * D_ + t * 16 + n0] = f32_bf16(o[t][r] / lrow[r]);
}

// ---------------------------------------------------------------------------
// RMSNorm: one block per row of 1024
// ---------------------------------------------------------------------------
__global__ __launch_bounds__(256) void rmsnorm(const float* __restrict__ in,
                                               const float* __restrict__ w,
                                               float* __restrict__ out) {
  __shared__ float red[8];
  int row = blockIdx.x;
  const float* r = in + (size_t)row * D_;
  float ss = 0.f;
  for (int d = threadIdx.x; d < D_; d += 256) { float v = r[d]; ss += v * v; }
#pragma unroll
  for (int m = 16; m; m >>= 1) ss += __shfl_xor(ss, m, 32);
  if ((threadIdx.x & 31) == 0) red[threadIdx.x >> 5] = ss;
  __syncthreads();
  float tot = 0.f;
#pragma unroll
  for (int i = 0; i < 8; ++i) tot += red[i];
  float inv = 1.0f / sqrtf(tot * (1.0f / D_) + 1e-6f);
  float* o = out + (size_t)row * D_;
  for (int d = threadIdx.x; d < D_; d += 256) o[d] = r[d] * inv * w[d];
}

// ---------------------------------------------------------------------------
extern "C" void kernel_launch(void* const* d_in, const int* in_sizes, int n_in,
                              void* d_out, int out_size, void* d_ws, size_t ws_size,
                              hipStream_t stream) {
  (void)in_sizes; (void)n_in; (void)out_size; (void)ws_size;
  const float* x  = (const float*)d_in[0];
  const float* Wq = (const float*)d_in[1];
  const float* Wk = (const float*)d_in[2];
  const float* Wv = (const float*)d_in[3];
  const float* Wo = (const float*)d_in[4];
  const float* nw = (const float*)d_in[5];
  float* out = (float*)d_out;

  const size_t MS  = (size_t)B_ * S_;          // 4096 rows
  const size_t KVW = (size_t)D_ * (HKV_ * HD_);

  u16*   Xb  = (u16*)d_ws;                     // x bf16             [MS, D]
  u16*   Wqt = Xb  + MS * D_;                  // Wq^T bf16          [D, D]
  u16*   Wkt = Wqt + (size_t)D_ * D_;          // Wk^T bf16          [256, D]
  u16*   Wvt = Wkt + KVW;                      // Wv^T bf16          [256, D]
  u16*   Wot = Wvt + KVW;                      // Wo^T bf16          [D, D]
  float* Qf  = (float*)(Wot + (size_t)D_ * D_);// q proj f32         [MS, D]
  float* Kf  = Qf + MS * D_;                   // k proj f32         [MS, 256]
  float* Vf  = Kf + MS * (HKV_ * HD_);         // v proj f32         [MS, 256]
  u16*   Qb  = (u16*)(Vf + MS * (HKV_ * HD_)); // roped Q bf16       [B,H,S,HD]
  u16*   Kb2 = Qb + MS * D_;                   // roped K bf16       [B,HKV,S,HD]
  u16*   Vb  = Kb2 + (size_t)B_ * HKV_ * S_ * HD_; // V^T bf16       [B,HKV,HD,S]
  u16*   Ab  = Vb  + (size_t)B_ * HKV_ * S_ * HD_; // attn out bf16  [MS, D]
  float* Of  = Qf;                             // reuse Qf for final f32 GEMM out

  dim3 tb(32, 8);
  cvt_bf16<<<2048, 256, 0, stream>>>(x, Xb, (int)(MS * D_));
  cvt_bf16_T<<<dim3(D_ / 32, D_ / 32), tb, 0, stream>>>(Wq, Wqt, D_, D_);
  cvt_bf16_T<<<dim3((HKV_ * HD_) / 32, D_ / 32), tb, 0, stream>>>(Wk, Wkt, D_, HKV_ * HD_);
  cvt_bf16_T<<<dim3((HKV_ * HD_) / 32, D_ / 32), tb, 0, stream>>>(Wv, Wvt, D_, HKV_ * HD_);
  cvt_bf16_T<<<dim3(D_ / 32, D_ / 32), tb, 0, stream>>>(Wo, Wot, D_, D_);

  dim3 g1((unsigned)(MS / BM), D_ / BN);
  dim3 g2((unsigned)(MS / BM), (HKV_ * HD_) / BN);
  gemm_bf16_f32<<<g1, 256, 0, stream>>>(Xb, Wqt, Qf, D_);
  gemm_bf16_f32<<<g2, 256, 0, stream>>>(Xb, Wkt, Kf, HKV_ * HD_);
  gemm_bf16_f32<<<g2, 256, 0, stream>>>(Xb, Wvt, Vf, HKV_ * HD_);

  int nq = B_ * H_ * S_ * HD_;
  int nk = B_ * HKV_ * S_ * HD_;
  rope_pack<<<nq / 256, 256, 0, stream>>>(Qf, Qb, H_, 0.125f);   // scale = HD^-0.5
  rope_pack<<<nk / 256, 256, 0, stream>>>(Kf, Kb2, HKV_, 1.0f);
  convert_v_T<<<nk / 256, 256, 0, stream>>>(Vf, Vb);

  attn_fwd<<<B_ * H_ * (S_ / 64), 128, 0, stream>>>(Qb, Kb2, Vb, Ab);

  gemm_bf16_f32<<<g1, 256, 0, stream>>>(Ab, Wot, Of, D_);
  rmsnorm<<<(unsigned)MS, 256, 0, stream>>>(Of, nw, out);
}
